// MultiHeadAttention_2705829397201
// MI455X (gfx1250) — matrix operand order repaired and checked by reading, not compiled, not run on hardware
//
#include <hip/hip_runtime.h>
#include <hip/hip_bf16.h>
#include <math.h>

#define D_MODEL 768
#define N_HEADS 12
#define D_KH    64
#define SEQ     4096

typedef __attribute__((ext_vector_type(8)))  _Float16 v8h;
typedef __attribute__((ext_vector_type(16))) _Float16 v16h;
typedef __attribute__((ext_vector_type(8)))  float    v8f;

static __device__ __forceinline__ int lane_id() { return (int)(threadIdx.x & 31); }

static __device__ __forceinline__ v16h make_v16(v8h a, v8h b) {
  v16h r;
#pragma unroll
  for (int i = 0; i < 8; ++i) { r[i] = a[i]; r[i + 8] = b[i]; }
  return r;
}

// A fragment (16x32 f16) from row-major matrix p[row][k], leading dim ld.
// ISA layout: lane(0-15)=row M, elems 0-7 -> K=k0+8*hi.., elems 8-15 -> K=k0+16+8*hi..
static __device__ __forceinline__ v16h load_a_frag(const _Float16* p, int ld, int row0, int k0) {
  const int lane = lane_id();
  const int hi = lane >> 4, m = lane & 15;
  const _Float16* base = p + (size_t)(row0 + m) * ld + (k0 + 8 * hi);
  v8h x = *(const v8h*)(base);
  v8h y = *(const v8h*)(base + 16);
  return make_v16(x, y);
}

// B fragment (32x16 f16) where B[k][n] = p[n][k] (p row-major, rows=N, cols=K).
// ISA layout: lane(0-15)=col N, elems 0-7 -> K=k0+8*hi.., elems 8-15 -> K=k0+16+8*hi..
static __device__ __forceinline__ v16h load_b_frag(const _Float16* p, int ld, int n0, int k0) {
  const int lane = lane_id();
  const int hi = lane >> 4, n = lane & 15;
  const _Float16* base = p + (size_t)(n0 + n) * ld + (k0 + 8 * hi);
  v8h x = *(const v8h*)(base);
  v8h y = *(const v8h*)(base + 16);
  return make_v16(x, y);
}

static __device__ __forceinline__ v8f wmma_f16(v16h a, v16h b, v8f c) {
  return __builtin_amdgcn_wmma_f32_16x16x32_f16(false, a, false, b, (short)0, c, false, false);
}

// ---------------------------------------------------------------------------
// fp32 -> f16 cast (optionally scaled; 1/sqrt(d_k) is folded into W_q here)
// ---------------------------------------------------------------------------
__global__ __launch_bounds__(256) void k_cvt_f16(const float* __restrict__ in,
                                                 _Float16* __restrict__ out,
                                                 int n, float scale) {
  int i = blockIdx.x * 256 + threadIdx.x;
  if (i < n) out[i] = (_Float16)(in[i] * scale);
}

// ---------------------------------------------------------------------------
// Out[m][n] = sum_k X[m][k] * Wt[n][k]   (i.e. X @ W^T), f16 in, f16 out.
// Wave tile 64x64 (4x4 WMMA tiles), workgroup 8 waves = 128x256.
// ---------------------------------------------------------------------------
__global__ __launch_bounds__(256) void k_gemm_xwT_f16(const _Float16* __restrict__ X,
                                                      const _Float16* __restrict__ Wt,
                                                      _Float16* __restrict__ Out,
                                                      int M, int N, int K) {
  const int wave = (int)(threadIdx.x >> 5);
  const int wm = wave >> 2, wn = wave & 3;
  const int m0 = blockIdx.y * 128 + wm * 64;
  const int n0 = blockIdx.x * 256 + wn * 64;
  const int lane = lane_id();
  const int hi = lane >> 4, ln = lane & 15;

  v8f c[4][4] = {};
  for (int k0 = 0; k0 < K; k0 += 32) {
    if (k0 + 64 < K) {
      __builtin_prefetch((const void*)(X + (size_t)(m0 + ln) * K + k0 + 64), 0, 1);
      __builtin_prefetch((const void*)(Wt + (size_t)(n0 + ln) * K + k0 + 64), 0, 1);
    }
    v16h a[4], b[4];
#pragma unroll
    for (int i = 0; i < 4; ++i) a[i] = load_a_frag(X, K, m0 + 16 * i, k0);
#pragma unroll
    for (int j = 0; j < 4; ++j) b[j] = load_b_frag(Wt, K, n0 + 16 * j, k0);
#pragma unroll
    for (int i = 0; i < 4; ++i)
#pragma unroll
      for (int j = 0; j < 4; ++j) c[i][j] = wmma_f16(a[i], b[j], c[i][j]);
  }
#pragma unroll
  for (int i = 0; i < 4; ++i)
#pragma unroll
    for (int j = 0; j < 4; ++j)
#pragma unroll
      for (int r = 0; r < 8; ++r)
        Out[(size_t)(m0 + 16 * i + r + 8 * hi) * N + (n0 + 16 * j + ln)] =
            (_Float16)c[i][j][r];
}

// ---------------------------------------------------------------------------
// Same GEMM but f32 output + bias (final output projection).
// ---------------------------------------------------------------------------
__global__ __launch_bounds__(256) void k_gemm_xwT_bias_f32(const _Float16* __restrict__ X,
                                                           const _Float16* __restrict__ Wt,
                                                           const float* __restrict__ bias,
                                                           float* __restrict__ Out,
                                                           int M, int N, int K) {
  const int wave = (int)(threadIdx.x >> 5);
  const int wm = wave >> 2, wn = wave & 3;
  const int m0 = blockIdx.y * 128 + wm * 64;
  const int n0 = blockIdx.x * 256 + wn * 64;
  const int lane = lane_id();
  const int hi = lane >> 4, ln = lane & 15;

  v8f c[4][4] = {};
  for (int k0 = 0; k0 < K; k0 += 32) {
    v16h a[4], b[4];
#pragma unroll
    for (int i = 0; i < 4; ++i) a[i] = load_a_frag(X, K, m0 + 16 * i, k0);
#pragma unroll
    for (int j = 0; j < 4; ++j) b[j] = load_b_frag(Wt, K, n0 + 16 * j, k0);
#pragma unroll
    for (int i = 0; i < 4; ++i)
#pragma unroll
      for (int j = 0; j < 4; ++j) c[i][j] = wmma_f16(a[i], b[j], c[i][j]);
  }
#pragma unroll
  for (int i = 0; i < 4; ++i)
#pragma unroll
    for (int j = 0; j < 4; ++j) {
      const int col = n0 + 16 * j + ln;
      const float bb = bias[col];
#pragma unroll
      for (int r = 0; r < 8; ++r)
        Out[(size_t)(m0 + 16 * i + r + 8 * hi) * N + col] = c[i][j][r] + bb;
    }
}

// ---------------------------------------------------------------------------
// Flash attention, one head per blockIdx.y. Each wave owns 16 query rows;
// workgroup (8 waves) streams 64-key K / V^T tiles through LDS with online
// softmax. Q was pre-scaled by 1/sqrt(d_k) (folded into W_q).
// H[q][head*64 + d] written as f16 (concatenated-heads layout).
// ---------------------------------------------------------------------------
__global__ __launch_bounds__(256) void k_flash_attn(const _Float16* __restrict__ Q,
                                                    const _Float16* __restrict__ Kc,
                                                    const _Float16* __restrict__ V,
                                                    _Float16* __restrict__ H) {
  __shared__ _Float16 sK[64][72];        // K tile, row-major [key][d], padded
  __shared__ _Float16 sVT[64][72];       // V tile transposed: [d][key]
  __shared__ _Float16 sP[8][16][72];     // per-wave softmax-probability scratch

  const int head = blockIdx.y;
  const int hc = head * D_KH;
  const int wave = (int)(threadIdx.x >> 5);
  const int lane = lane_id();
  const int hi = lane >> 4, ln = lane & 15;
  const int q0 = blockIdx.x * 128 + wave * 16;
  const int t = (int)threadIdx.x;

  // Q fragments for this wave's 16 rows (K-depth 64 = two 16x32 A frags)
  const v16h qa0 = load_a_frag(Q, D_MODEL, q0, hc + 0);
  const v16h qa1 = load_a_frag(Q, D_MODEL, q0, hc + 32);

  v8f o[4] = {};                 // O accumulator: 16 rows x 64 cols
  float mrow[8], lrow[8];
#pragma unroll
  for (int i = 0; i < 8; ++i) { mrow[i] = -INFINITY; lrow[i] = 0.0f; }

  for (int kv0 = 0; kv0 < SEQ; kv0 += 64) {
    __syncthreads();
    {  // stage K tile: thread -> (row, 16-col slice)
      const int row = t >> 2, part = t & 3;
      const _Float16* src = Kc + (size_t)(kv0 + row) * D_MODEL + hc + part * 16;
      v8h x = *(const v8h*)src;
      v8h y = *(const v8h*)(src + 8);
      *(v8h*)&sK[row][part * 16] = x;
      *(v8h*)&sK[row][part * 16 + 8] = y;
    }
    {  // stage V transposed: thread -> (key, 16 d values scattered)
      const int key = t & 63, part = t >> 6;
      const _Float16* src = V + (size_t)(kv0 + key) * D_MODEL + hc + part * 16;
      v8h x = *(const v8h*)src;
      v8h y = *(const v8h*)(src + 8);
#pragma unroll
      for (int j = 0; j < 8; ++j) {
        sVT[part * 16 + j][key] = x[j];
        sVT[part * 16 + 8 + j][key] = y[j];
      }
    }
    __syncthreads();

    // scores S = Q_tile @ K_tile^T  (4 key subtiles of 16)
    v8f s[4];
#pragma unroll
    for (int nt = 0; nt < 4; ++nt) {
      v8f z = {};
      z = wmma_f16(qa0, load_b_frag(&sK[0][0], 72, nt * 16, 0), z);
      z = wmma_f16(qa1, load_b_frag(&sK[0][0], 72, nt * 16, 32), z);
      s[nt] = z;
    }

    // online softmax: row max / rescale / exp / row sum
    float mnew[8], sf[8], rsum[8];
#pragma unroll
    for (int i = 0; i < 8; ++i) {
      float v = s[0][i];
      v = fmaxf(v, s[1][i]); v = fmaxf(v, s[2][i]); v = fmaxf(v, s[3][i]);
#pragma unroll
      for (int off = 1; off < 16; off <<= 1) v = fmaxf(v, __shfl_xor(v, off, 32));
      mnew[i] = fmaxf(mrow[i], v);
      sf[i] = __expf(mrow[i] - mnew[i]);
      mrow[i] = mnew[i];
      rsum[i] = 0.0f;
    }
#pragma unroll
    for (int nt = 0; nt < 4; ++nt)
#pragma unroll
      for (int i = 0; i < 8; ++i) {
        float p = __expf(s[nt][i] - mnew[i]);
        rsum[i] += p;
        sP[wave][i + 8 * hi][nt * 16 + ln] = (_Float16)p;  // C-layout -> LDS
      }
#pragma unroll
    for (int i = 0; i < 8; ++i) {
      float v = rsum[i];
#pragma unroll
      for (int off = 1; off < 16; off <<= 1) v += __shfl_xor(v, off, 32);
      lrow[i] = lrow[i] * sf[i] + v;
    }
#pragma unroll
    for (int dt = 0; dt < 4; ++dt)
#pragma unroll
      for (int i = 0; i < 8; ++i) o[dt][i] *= sf[i];

    // O += P @ V  (P re-read from LDS as A frags; V^T gives contiguous B frags)
    const v16h pa0 = load_a_frag(&sP[wave][0][0], 72, 0, 0);
    const v16h pa1 = load_a_frag(&sP[wave][0][0], 72, 0, 32);
#pragma unroll
    for (int dt = 0; dt < 4; ++dt) {
      o[dt] = wmma_f16(pa0, load_b_frag(&sVT[0][0], 72, dt * 16, 0), o[dt]);
      o[dt] = wmma_f16(pa1, load_b_frag(&sVT[0][0], 72, dt * 16, 32), o[dt]);
    }
  }

  // normalize and write concatenated-heads H (f16)
#pragma unroll
  for (int dt = 0; dt < 4; ++dt)
#pragma unroll
    for (int i = 0; i < 8; ++i) {
      float v = o[dt][i] / lrow[i];
      H[(size_t)(q0 + i + 8 * hi) * D_MODEL + hc + dt * 16 + ln] = (_Float16)v;
    }
}

// ---------------------------------------------------------------------------
extern "C" void kernel_launch(void* const* d_in, const int* in_sizes, int n_in,
                              void* d_out, int out_size, void* d_ws, size_t ws_size,
                              hipStream_t stream) {
  (void)in_sizes; (void)n_in; (void)out_size; (void)ws_size;

  const float* x_q = (const float*)d_in[0];
  const float* x_k = (const float*)d_in[1];
  const float* x_v = (const float*)d_in[2];
  const float* w_q = (const float*)d_in[3];
  const float* w_k = (const float*)d_in[4];
  const float* w_v = (const float*)d_in[5];
  const float* w_h = (const float*)d_in[6];
  const float* b_h = (const float*)d_in[7];
  float* out = (float*)d_out;

  const size_t SX = (size_t)SEQ * D_MODEL;       // 3,145,728
  const size_t SW = (size_t)D_MODEL * D_MODEL;   //   589,824

  _Float16* p = (_Float16*)d_ws;
  _Float16* Xq = p; p += SX;
  _Float16* Xk = p; p += SX;
  _Float16* Xv = p; p += SX;
  _Float16* Wq = p; p += SW;
  _Float16* Wk = p; p += SW;
  _Float16* Wv = p; p += SW;
  _Float16* Wh = p; p += SW;
  _Float16* Qm = p; p += SX;
  _Float16* Km = p; p += SX;
  _Float16* Vm = p; p += SX;
  _Float16* Hm = p; p += SX;   // total ~48.8 MB of workspace

  const int nX = (int)SX, nW = (int)SW;
  const int gX = (nX + 255) / 256, gW = (nW + 255) / 256;
  const float qscale = 1.0f / 8.0f;  // 1/sqrt(d_k), folded into W_q

  k_cvt_f16<<<gX, 256, 0, stream>>>(x_q, Xq, nX, 1.0f);
  k_cvt_f16<<<gX, 256, 0, stream>>>(x_k, Xk, nX, 1.0f);
  k_cvt_f16<<<gX, 256, 0, stream>>>(x_v, Xv, nX, 1.0f);
  k_cvt_f16<<<gW, 256, 0, stream>>>(w_q, Wq, nW, qscale);
  k_cvt_f16<<<gW, 256, 0, stream>>>(w_k, Wk, nW, 1.0f);
  k_cvt_f16<<<gW, 256, 0, stream>>>(w_v, Wv, nW, 1.0f);
  k_cvt_f16<<<gW, 256, 0, stream>>>(w_h, Wh, nW, 1.0f);

  dim3 gp(D_MODEL / 256, SEQ / 128);   // (3, 32)
  k_gemm_xwT_f16<<<gp, 256, 0, stream>>>(Xq, Wq, Qm, SEQ, D_MODEL, D_MODEL);
  k_gemm_xwT_f16<<<gp, 256, 0, stream>>>(Xk, Wk, Km, SEQ, D_MODEL, D_MODEL);
  k_gemm_xwT_f16<<<gp, 256, 0, stream>>>(Xv, Wv, Vm, SEQ, D_MODEL, D_MODEL);

  dim3 ga(SEQ / 128, N_HEADS);         // (32, 12)
  k_flash_attn<<<ga, 256, 0, stream>>>(Qm, Km, Vm, Hm);

  k_gemm_xwT_bias_f32<<<gp, 256, 0, stream>>>(Hm, Wh, b_h, out, SEQ, D_MODEL, D_MODEL);
}
